// TransformerFlowReconstructionModel_30906584662603
// MI455X (gfx1250) — matrix operand order repaired and checked by reading, not compiled, not run on hardware
//
#include <hip/hip_runtime.h>

// ---------------------------------------------------------------------------
// Transformer (s==1 => attention == V-projection) as a chain of bf16 WMMA
// GEMMs with fused bias/residual/ReLU/LayerNorm epilogues. gfx1250 wave32.
// ---------------------------------------------------------------------------

#define BSZ  4096
#define DIN  8192
#define DM   512
#define DFF  2048
#define DOUT 8192

typedef __attribute__((ext_vector_type(16))) __bf16 v16bf;
typedef __attribute__((ext_vector_type(8)))  __bf16 v8bf;
typedef __attribute__((ext_vector_type(8)))  float  v8f;

union V16 { v16bf v; v8bf h[2]; };

// ------------------------------- prep kernels ------------------------------

__global__ void k_cvt_bf16(const float* __restrict__ in, __bf16* __restrict__ out,
                           long long n) {
  long long i  = (long long)blockIdx.x * blockDim.x + threadIdx.x;
  long long st = (long long)gridDim.x * blockDim.x;
  for (; i < n; i += st) out[i] = (__bf16)in[i];
}

// W (K x N, row-major f32)  ->  Wt (N x K, row-major bf16):  Wt[n*K+k] = W[k*N+n]
__global__ void k_transpose_bf16(const float* __restrict__ W, __bf16* __restrict__ Wt,
                                 int K, int N) {
  long long total = (long long)K * N;
  long long i  = (long long)blockIdx.x * blockDim.x + threadIdx.x;
  long long st = (long long)gridDim.x * blockDim.x;
  for (; i < total; i += st) {
    int n = (int)(i / K), k = (int)(i % K);
    Wt[i] = (__bf16)W[(long long)k * N + n];
  }
}

// ------------------------------- GEMM kernel -------------------------------
// MODE 0: embed   : out = A@B + bias + time_emb[t[r]] + pos        -> f32+bf16
// MODE 1: LN step : out = LayerNorm(resid + A@B + bias; g,b)       -> f32+bf16
// MODE 2: FF1     : out = relu(A@B + bias)                         -> bf16
// MODE 3: final   : out = A@B + bias                               -> f32
//
// Block: 256 threads = 8 waves.  Workgroup tile 32(M) x 512(N); wave w owns
// cols [w*64, w*64+64) => 2x4 = 8 accumulator tiles, K stepped by 32.
template<int MODE>
__global__ __launch_bounds__(256)
void k_gemm(const __bf16* __restrict__ A, const __bf16* __restrict__ Bt,
            const float* __restrict__ bias, const float* __restrict__ resid,
            const float* __restrict__ lng, const float* __restrict__ lnb,
            const float* __restrict__ temb, const int* __restrict__ tvec,
            const float* __restrict__ pos,
            float* __restrict__ outF, __bf16* __restrict__ outB,
            int M, int N, int K)
{
  const int lane = threadIdx.x & 31;
  const int wv   = threadIdx.x >> 5;
  const int hl   = lane >> 4;        // half-wave select
  const int l16  = lane & 15;
  const int m0   = blockIdx.x * 32;
  const int n0   = blockIdx.y * 512 + wv * 64;

  v8f acc[2][4] = {};

  // A fragment base: 16 contiguous K per lane (lane<16: K 0..15; lane>=16: K 16..31)
  const __bf16* a0p = A + (size_t)(m0 + l16) * K + hl * 16;
  const __bf16* a1p = a0p + (size_t)16 * K;
  // B fragment base: Bt row = output column, 16 contiguous K per lane
  const __bf16* bp0 = Bt + (size_t)(n0 +  0 + l16) * K + hl * 16;
  const __bf16* bp1 = Bt + (size_t)(n0 + 16 + l16) * K + hl * 16;
  const __bf16* bp2 = Bt + (size_t)(n0 + 32 + l16) * K + hl * 16;
  const __bf16* bp3 = Bt + (size_t)(n0 + 48 + l16) * K + hl * 16;

  for (int k0 = 0; k0 < K; k0 += 32) {
    V16 a0, a1;
    a0.h[0] = *(const v8bf*)(a0p + k0);
    a0.h[1] = *(const v8bf*)(a0p + k0 + 8);
    a1.h[0] = *(const v8bf*)(a1p + k0);
    a1.h[1] = *(const v8bf*)(a1p + k0 + 8);
    v16bf b0 = *(const v16bf*)(bp0 + k0);
    v16bf b1 = *(const v16bf*)(bp1 + k0);
    v16bf b2 = *(const v16bf*)(bp2 + k0);
    v16bf b3 = *(const v16bf*)(bp3 + k0);

    acc[0][0] = __builtin_amdgcn_wmma_f32_16x16x32_bf16(false, a0.v, false, b0, (short)0, acc[0][0], false, false);
    acc[0][1] = __builtin_amdgcn_wmma_f32_16x16x32_bf16(false, a0.v, false, b1, (short)0, acc[0][1], false, false);
    acc[0][2] = __builtin_amdgcn_wmma_f32_16x16x32_bf16(false, a0.v, false, b2, (short)0, acc[0][2], false, false);
    acc[0][3] = __builtin_amdgcn_wmma_f32_16x16x32_bf16(false, a0.v, false, b3, (short)0, acc[0][3], false, false);
    acc[1][0] = __builtin_amdgcn_wmma_f32_16x16x32_bf16(false, a1.v, false, b0, (short)0, acc[1][0], false, false);
    acc[1][1] = __builtin_amdgcn_wmma_f32_16x16x32_bf16(false, a1.v, false, b1, (short)0, acc[1][1], false, false);
    acc[1][2] = __builtin_amdgcn_wmma_f32_16x16x32_bf16(false, a1.v, false, b2, (short)0, acc[1][2], false, false);
    acc[1][3] = __builtin_amdgcn_wmma_f32_16x16x32_bf16(false, a1.v, false, b3, (short)0, acc[1][3], false, false);
  }

  // D tile layout: VGPR v, lanes 0-15 -> row v, lanes 16-31 -> row v+8; col = lane%16.
  if constexpr (MODE == 1) {
    __shared__ float sw1[8][32], sw2[8][32], smean[32], srstd[32];
    // residual + bias, per-wave partial row sums (deterministic)
    #pragma unroll
    for (int mi = 0; mi < 2; ++mi) {
      #pragma unroll
      for (int v = 0; v < 8; ++v) {
        const int lr = mi * 16 + v + 8 * hl;   // local row 0..31
        const int r  = m0 + lr;
        float ps = 0.f, ps2 = 0.f;
        #pragma unroll
        for (int ni = 0; ni < 4; ++ni) {
          const int c = n0 + ni * 16 + l16;
          float t = acc[mi][ni][v] + bias[c] + resid[(size_t)r * DM + c];
          acc[mi][ni][v] = t;
          ps += t; ps2 += t * t;
        }
        // reduce across the 16 lanes of this half-wave (all hold the same row)
        #pragma unroll
        for (int d = 1; d < 16; d <<= 1) {
          ps  += __shfl_xor(ps,  d, 32);
          ps2 += __shfl_xor(ps2, d, 32);
        }
        if (l16 == 0) { sw1[wv][lr] = ps; sw2[wv][lr] = ps2; }
      }
    }
    __syncthreads();
    if (threadIdx.x < 32) {
      float s = 0.f, q = 0.f;
      #pragma unroll
      for (int w = 0; w < 8; ++w) { s += sw1[w][threadIdx.x]; q += sw2[w][threadIdx.x]; }
      const float mn = s * (1.0f / 512.0f);
      const float vr = q * (1.0f / 512.0f) - mn * mn;
      smean[threadIdx.x] = mn;
      srstd[threadIdx.x] = rsqrtf(vr + 1e-5f);
    }
    __syncthreads();
    #pragma unroll
    for (int mi = 0; mi < 2; ++mi)
      #pragma unroll
      for (int ni = 0; ni < 4; ++ni)
        #pragma unroll
        for (int v = 0; v < 8; ++v) {
          const int lr = mi * 16 + v + 8 * hl;
          const int r  = m0 + lr;
          const int c  = n0 + ni * 16 + l16;
          float o = (acc[mi][ni][v] - smean[lr]) * srstd[lr] * lng[c] + lnb[c];
          outF[(size_t)r * DM + c] = o;
          outB[(size_t)r * DM + c] = (__bf16)o;
        }
  } else {
    #pragma unroll
    for (int mi = 0; mi < 2; ++mi)
      #pragma unroll
      for (int ni = 0; ni < 4; ++ni)
        #pragma unroll
        for (int v = 0; v < 8; ++v) {
          const int r = m0 + mi * 16 + v + 8 * hl;
          const int c = n0 + ni * 16 + l16;
          float val = acc[mi][ni][v] + bias[c];
          if constexpr (MODE == 0) {
            val += temb[(size_t)tvec[r] * DM + c] + pos[c];
            outF[(size_t)r * N + c] = val;
            outB[(size_t)r * N + c] = (__bf16)val;
          } else if constexpr (MODE == 2) {
            val = fmaxf(val, 0.f);
            outB[(size_t)r * N + c] = (__bf16)val;
          } else {
            outF[(size_t)r * N + c] = val;
          }
        }
  }
}

// --------------------------------- driver ----------------------------------
// Input order: setup_inputs() dict insertion order, recursively flattened:
//   0:x 1:t 2:embed.W 3:embed.b 4:time_emb 5:pos
//   enc layer i @ 6+15i : qW qb kW kb vW vb rel l1W l1b l2W l2b ln1g ln1b ln2g ln2b
//   dec layer i @ 96+24i: saqW saqb sakW sakb savW savb sarel
//                         caqW caqb cakW cakb cavW cavb carel
//                         l1W l1b l2W l2b ln1g ln1b ln2g ln2b ln3g ln3b
//   240:out.W 241:out.b
extern "C" void kernel_launch(void* const* d_in, const int* in_sizes, int n_in,
                              void* d_out, int out_size, void* d_ws, size_t ws_size,
                              hipStream_t stream) {
  (void)in_sizes; (void)n_in; (void)out_size; (void)ws_size;
  auto F = [&](int i) { return (const float*)d_in[i]; };
  const float* x    = (const float*)d_in[0];
  const int*   tv   = (const int*)d_in[1];
  const float* embW = F(2);
  const float* embB = F(3);
  const float* temb = F(4);
  const float* pos  = F(5);
  const float* outW = F(240);
  const float* outB = F(241);

  // workspace carving (256B aligned)
  char* p = (char*)d_ws;
  auto alloc = [&](size_t bytes) -> void* {
    void* r = (void*)p;
    p += (bytes + 255) & ~(size_t)255;
    return r;
  };
  __bf16* xbf   = (__bf16*)alloc((size_t)BSZ * DIN * 2);
  float*  hf[2]; hf[0] = (float*)alloc((size_t)BSZ * DM * 4);
                 hf[1] = (float*)alloc((size_t)BSZ * DM * 4);
  __bf16* hb[2]; hb[0] = (__bf16*)alloc((size_t)BSZ * DM * 2);
                 hb[1] = (__bf16*)alloc((size_t)BSZ * DM * 2);
  __bf16* ffb   = (__bf16*)alloc((size_t)BSZ * DFF * 2);
  __bf16* embT  = (__bf16*)alloc((size_t)DIN * DM * 2);
  __bf16* outT  = (__bf16*)alloc((size_t)DM * DOUT * 2);
  __bf16 *eV[6], *eL1[6], *eL2[6], *dSA[6], *dCA[6], *dL1[6], *dL2[6];
  for (int i = 0; i < 6; ++i) {
    eV[i]  = (__bf16*)alloc((size_t)DM * DM * 2);
    eL1[i] = (__bf16*)alloc((size_t)DM * DFF * 2);
    eL2[i] = (__bf16*)alloc((size_t)DFF * DM * 2);
    dSA[i] = (__bf16*)alloc((size_t)DM * DM * 2);
    dCA[i] = (__bf16*)alloc((size_t)DM * DM * 2);
    dL1[i] = (__bf16*)alloc((size_t)DM * DFF * 2);
    dL2[i] = (__bf16*)alloc((size_t)DFF * DM * 2);
  }

  // --- prep: bf16 conversion of activations, transposed bf16 weights ---
  k_cvt_bf16<<<8192, 256, 0, stream>>>(x, xbf, (long long)BSZ * DIN);
  auto tr = [&](const float* W, __bf16* T, int K, int N) {
    long long tot = (long long)K * N;
    int blocks = (int)((tot + 255) / 256);
    if (blocks > 16384) blocks = 16384;
    k_transpose_bf16<<<blocks, 256, 0, stream>>>(W, T, K, N);
  };
  tr(embW, embT, DIN, DM);
  tr(outW, outT, DM, DOUT);
  for (int i = 0; i < 6; ++i) {
    const int eb = 6 + 15 * i;
    tr(F(eb + 4), eV[i],  DM, DM);
    tr(F(eb + 7), eL1[i], DM, DFF);
    tr(F(eb + 9), eL2[i], DFF, DM);
    const int db = 96 + 24 * i;
    tr(F(db + 4),  dSA[i], DM, DM);
    tr(F(db + 11), dCA[i], DM, DM);
    tr(F(db + 14), dL1[i], DM, DFF);
    tr(F(db + 16), dL2[i], DFF, DM);
  }

  // --- embed: e = x@Wemb + b + time_emb[t] + pos ---
  k_gemm<0><<<dim3(BSZ / 32, 1), 256, 0, stream>>>(
      xbf, embT, embB, nullptr, nullptr, nullptr, temb, tv, pos,
      hf[0], hb[0], BSZ, DM, DIN);
  int cur = 0;

  // --- encoder ---
  for (int i = 0; i < 6; ++i) {
    const int eb = 6 + 15 * i;
    // h = LN1(h + h@Wv + bv)
    k_gemm<1><<<dim3(BSZ / 32, 1), 256, 0, stream>>>(
        hb[cur], eV[i], F(eb + 5), hf[cur], F(eb + 11), F(eb + 12),
        nullptr, nullptr, nullptr, hf[1 - cur], hb[1 - cur], BSZ, DM, DM);
    cur ^= 1;
    // ff = relu(h@W1 + b1)
    k_gemm<2><<<dim3(BSZ / 32, DFF / 512), 256, 0, stream>>>(
        hb[cur], eL1[i], F(eb + 8), nullptr, nullptr, nullptr,
        nullptr, nullptr, nullptr, nullptr, ffb, BSZ, DFF, DM);
    // h = LN2(h + ff@W2 + b2)
    k_gemm<1><<<dim3(BSZ / 32, 1), 256, 0, stream>>>(
        ffb, eL2[i], F(eb + 10), hf[cur], F(eb + 13), F(eb + 14),
        nullptr, nullptr, nullptr, hf[1 - cur], hb[1 - cur], BSZ, DM, DFF);
    cur ^= 1;
  }

  // --- decoder ---
  for (int i = 0; i < 6; ++i) {
    const int db = 96 + 24 * i;
    // g = LN1(g + g@Wv_sa + b)
    k_gemm<1><<<dim3(BSZ / 32, 1), 256, 0, stream>>>(
        hb[cur], dSA[i], F(db + 5), hf[cur], F(db + 18), F(db + 19),
        nullptr, nullptr, nullptr, hf[1 - cur], hb[1 - cur], BSZ, DM, DM);
    cur ^= 1;
    // g = LN2(g + g@Wv_ca + b)
    k_gemm<1><<<dim3(BSZ / 32, 1), 256, 0, stream>>>(
        hb[cur], dCA[i], F(db + 12), hf[cur], F(db + 20), F(db + 21),
        nullptr, nullptr, nullptr, hf[1 - cur], hb[1 - cur], BSZ, DM, DM);
    cur ^= 1;
    // ff
    k_gemm<2><<<dim3(BSZ / 32, DFF / 512), 256, 0, stream>>>(
        hb[cur], dL1[i], F(db + 15), nullptr, nullptr, nullptr,
        nullptr, nullptr, nullptr, nullptr, ffb, BSZ, DFF, DM);
    k_gemm<1><<<dim3(BSZ / 32, 1), 256, 0, stream>>>(
        ffb, dL2[i], F(db + 17), hf[cur], F(db + 22), F(db + 23),
        nullptr, nullptr, nullptr, hf[1 - cur], hb[1 - cur], BSZ, DM, DFF);
    cur ^= 1;
  }

  // --- output projection (g.mean(axis=1) == g for s==1) ---
  k_gemm<3><<<dim3(BSZ / 32, DOUT / 512), 256, 0, stream>>>(
      hb[cur], outT, outB, nullptr, nullptr, nullptr,
      nullptr, nullptr, nullptr, (float*)d_out, nullptr, BSZ, DOUT, DM);
}
